// Net_10617159156373
// MI455X (gfx1250) — matrix operand order repaired and checked by reading, not compiled, not run on hardware
//
#include <hip/hip_runtime.h>
#include <hip/hip_bf16.h>
#include <math.h>

typedef __attribute__((ext_vector_type(16))) _Float16 v16h;
typedef __attribute__((ext_vector_type(8)))  float    v8f;

#define BN_SCALE 0.9999950000374997f   /* 1/sqrt(1+1e-5), eval-mode BN fold */

// ---------------------------------------------------------------------------
// WMMA fragment helpers (CDNA5 wave32, ISA 7.12.2 layouts)
// A: 16x32 f16. lane L: row = L&15, half = L>>4.
//    VGPR j (pair 2j,2j+1): k = (j>>2)*16 + half*8 + (j&3)*2  (+k0)
//    -> per lane: halves [k0+half*8 .. +7] and [k0+16+half*8 .. +7]
//       i.e. two contiguous 16B runs => 2x (ds|global)_load_b128
// B: 32x16 f16. lane L: col = L&15. VGPR j: same k as A (+k0)
//    -> pre-swizzled fragment-major so each lane reads 32 contiguous bytes
// D: 16x16 f32. VGPR r: row = r + half*8, col = L&15
// ---------------------------------------------------------------------------
__device__ __forceinline__ v16h load_a_rowmajor(const _Float16* __restrict__ buf,
                                                int ld, int row0, int k0) {
  int lane = threadIdx.x & 31;
  int half = lane >> 4;
  const _Float16* p = buf + (row0 + (lane & 15)) * ld + k0;
  v16h a;
#pragma unroll
  for (int j = 0; j < 8; ++j) {
    int k = ((j >> 2) << 4) + (half << 3) + ((j & 3) << 1);
    a[2 * j]     = p[k];
    a[2 * j + 1] = p[k + 1];
  }
  return a;
}

// Packed weights: Wp laid out as [kc][nt][lane 0..31][16 halves] (512 halves
// per 16x16x32 fragment). Each lane's fragment is one contiguous 32B vector.
__device__ __forceinline__ v16h load_b_packed(const _Float16* __restrict__ Wp,
                                              int NT, int kc, int nt) {
  int lane = threadIdx.x & 31;
  const _Float16* p = Wp + (((size_t)(kc * NT + nt)) << 9) + (lane << 4);
  return *(const v16h*)p;
}

__device__ __forceinline__ v8f wmma_f16(v16h a, v16h b, v8f c) {
  return __builtin_amdgcn_wmma_f32_16x16x32_f16(false, a, false, b, (short)0, c,
                                                false, false);
}

// ---------------------------------------------------------------------------
// Weight convert to f16, zero-pad K to multiple of 32, and swizzle into
// B-fragment-major layout:
//   dst[((kc*(N/16)+nt)*32 + lane)*16 + e] = W[(kc*32 + (e>>3)*16 + (lane>>4)*8
//                                              + (e&7))*N + nt*16 + (lane&15)]
// ---------------------------------------------------------------------------
__global__ void pack_b_kernel(const float* __restrict__ src,
                              _Float16* __restrict__ dst,
                              int K, int N, int KPAD) {
  int idx = blockIdx.x * blockDim.x + threadIdx.x;
  if (idx >= KPAD * N) return;
  int e    = idx & 15;
  int lane = (idx >> 4) & 31;
  int frag = idx >> 9;
  int NT = N >> 4;
  int nt = frag % NT, kc = frag / NT;
  int k = kc * 32 + (e >> 3) * 16 + (lane >> 4) * 8 + (e & 7);
  int n = nt * 16 + (lane & 15);
  dst[idx] = (k < K) ? (_Float16)src[k * N + n] : (_Float16)0.0f;
}

// ---------------------------------------------------------------------------
// Farthest point sampling: 1 block per cloud, blockDim.x == n
// ---------------------------------------------------------------------------
__global__ void fps_kernel(const float* __restrict__ pos, int n, int M,
                           int* __restrict__ idx_out, float* __restrict__ ctr_out) {
  int b = blockIdx.x, t = threadIdx.x;
  const float* P = pos + (size_t)b * n * 3;
  __shared__ float swv[32];
  __shared__ int   swi[32];
  __shared__ int   s_next;
  float px = P[t * 3 + 0], py = P[t * 3 + 1], pz = P[t * 3 + 2];
  float mind = 1e10f;
  if (t == 0) {
    idx_out[b * M] = 0;
    ctr_out[(b * M) * 3 + 0] = P[0];
    ctr_out[(b * M) * 3 + 1] = P[1];
    ctr_out[(b * M) * 3 + 2] = P[2];
  }
  int last = 0;
  int nw = blockDim.x >> 5;
  for (int m = 1; m < M; ++m) {
    float lx = P[last * 3 + 0], ly = P[last * 3 + 1], lz = P[last * 3 + 2];
    float dx = px - lx, dy = py - ly, dz = pz - lz;
    mind = fminf(mind, dx * dx + dy * dy + dz * dz);
    float v = mind;
    int   i = t;
#pragma unroll
    for (int off = 16; off > 0; off >>= 1) {       // argmax, first-index ties
      float ov = __shfl_down(v, off, 32);
      int   oi = __shfl_down(i, off, 32);
      if (ov > v || (ov == v && oi < i)) { v = ov; i = oi; }
    }
    if ((t & 31) == 0) { swv[t >> 5] = v; swi[t >> 5] = i; }
    __syncthreads();
    if (t < 32) {
      float vv = (t < nw) ? swv[t] : -1e30f;
      int   ii = (t < nw) ? swi[t] : 0x7fffffff;
#pragma unroll
      for (int off = 16; off > 0; off >>= 1) {
        float ov = __shfl_down(vv, off, 32);
        int   oi = __shfl_down(ii, off, 32);
        if (ov > vv || (ov == vv && oi < ii)) { vv = ov; ii = oi; }
      }
      if (t == 0) s_next = ii;
    }
    __syncthreads();
    last = s_next;
    if (t == 0) {
      idx_out[b * M + m] = last;
      ctr_out[(b * M + m) * 3 + 0] = P[last * 3 + 0];
      ctr_out[(b * M + m) * 3 + 1] = P[last * 3 + 1];
      ctr_out[(b * M + m) * 3 + 2] = P[last * 3 + 2];
    }
  }
}

// ---------------------------------------------------------------------------
// 64 nearest neighbors per centroid + radius mask. 1 block (128 thr)/centroid.
// ---------------------------------------------------------------------------
__global__ void knn_kernel(const float* __restrict__ pts, int n,
                           const float* __restrict__ ctr, int M, float r2,
                           int* __restrict__ nbr,
                           unsigned long long* __restrict__ mask) {
  __shared__ float sd[1024];
  __shared__ float swv[4];
  __shared__ int   swi[4];
  int bm = blockIdx.x;
  int b  = bm / M;
  int tid = threadIdx.x;
  float cx = ctr[bm * 3 + 0], cy = ctr[bm * 3 + 1], cz = ctr[bm * 3 + 2];
  const float* P = pts + (size_t)b * n * 3;
  for (int i = tid; i < n; i += 128) {
    float dx = P[i * 3 + 0] - cx, dy = P[i * 3 + 1] - cy, dz = P[i * 3 + 2] - cz;
    sd[i] = dx * dx + dy * dy + dz * dz;
  }
  __syncthreads();
  for (int j = 0; j < 64; ++j) {
    float v = 3.4e38f;
    int   idx = 0x7fffffff;
    for (int i = tid; i < n; i += 128) {
      float d = sd[i];
      if (d < v || (d == v && i < idx)) { v = d; idx = i; }
    }
#pragma unroll
    for (int off = 16; off > 0; off >>= 1) {
      float ov = __shfl_down(v, off, 32);
      int   oi = __shfl_down(idx, off, 32);
      if (ov < v || (ov == v && oi < idx)) { v = ov; idx = oi; }
    }
    if ((tid & 31) == 0) { swv[tid >> 5] = v; swi[tid >> 5] = idx; }
    __syncthreads();
    if (tid == 0) {
      float bv = swv[0]; int bi = swi[0];
#pragma unroll
      for (int w = 1; w < 4; ++w) {
        if (swv[w] < bv || (swv[w] == bv && swi[w] < bi)) { bv = swv[w]; bi = swi[w]; }
      }
      nbr[bm * 64 + j] = bi;
      sd[bi] = 3.4e38f;
      if (j == 0) mask[bm] = 0ull;
      if (bv < r2) mask[bm] |= (1ull << j);
    }
    __syncthreads();
  }
}

// ---------------------------------------------------------------------------
// Fused SA module: gather features -> 3-layer WMMA MLP -> masked max-pool.
// 1 block (4 waves) per centroid; 64 neighbor rows.
// ---------------------------------------------------------------------------
template <int CFEAT, int CIN_PAD, int C1, int C2, int C3>
__global__ void __launch_bounds__(128)
sa_kernel(const float* __restrict__ P, const float* __restrict__ feat,
          const float* __restrict__ ctr, const int* __restrict__ nbr,
          const unsigned long long* __restrict__ mask, int n_src, int M,
          const _Float16* __restrict__ W0, const _Float16* __restrict__ W1,
          const _Float16* __restrict__ W2,
          const float* __restrict__ b0, const float* __restrict__ g0,
          const float* __restrict__ be0,
          const float* __restrict__ b1, const float* __restrict__ g1,
          const float* __restrict__ be1,
          const float* __restrict__ b2, float* __restrict__ xout) {
  constexpr int CA = (CIN_PAD > C2) ? CIN_PAD : C2;
  __shared__ _Float16 bufA[64 * CA];
  __shared__ _Float16 bufB[64 * C1];
  __shared__ float    pool[4 * C3];

  int bm = blockIdx.x;
  int b  = bm / M;
  int tid = threadIdx.x;
  const int* nb = nbr + bm * 64;
  float cx = ctr[bm * 3 + 0], cy = ctr[bm * 3 + 1], cz = ctr[bm * 3 + 2];

  // stage input rows: [prev features | rel pos | zero pad] as f16
  for (int i = tid; i < 64 * CIN_PAD; i += 128) {
    int row = i / CIN_PAD, col = i % CIN_PAD;
    int q = b * n_src + nb[row];
    float v;
    if (col < CFEAT) {
      v = feat[q * CFEAT + col];
    } else if (col < CFEAT + 3) {
      int d = col - CFEAT;
      float c = (d == 0) ? cx : ((d == 1) ? cy : cz);
      v = P[q * 3 + d] - c;
    } else {
      v = 0.0f;
    }
    bufA[row * CA + col] = (_Float16)v;
  }
  __syncthreads();

  int wave = tid >> 5, lane = tid & 31, half = lane >> 4, nl = lane & 15;
  int row0 = wave * 16;

  // layer 0: bufA(K=CIN_PAD) -> bufB(C1), bias+BN+ReLU
  for (int ct = 0; ct < C1 / 16; ++ct) {
    v8f acc = {};
    for (int kc = 0; kc < CIN_PAD / 32; ++kc) {
      v16h a = load_a_rowmajor(bufA, CA, row0, kc * 32);
      v16h w = load_b_packed(W0, C1 / 16, kc, ct);
      acc = wmma_f16(a, w, acc);
    }
    int n = ct * 16 + nl;
    float bb = b0[n], sc = g0[n] * BN_SCALE, bt = be0[n];
#pragma unroll
    for (int r = 0; r < 8; ++r) {
      int mm = row0 + r + half * 8;
      float v = (acc[r] + bb) * sc + bt;
      bufB[mm * C1 + n] = (_Float16)(v > 0.0f ? v : 0.0f);
    }
  }
  __syncthreads();

  // layer 1: bufB(K=C1) -> bufA(C2), bias+BN+ReLU
  for (int ct = 0; ct < C2 / 16; ++ct) {
    v8f acc = {};
    for (int kc = 0; kc < C1 / 32; ++kc) {
      v16h a = load_a_rowmajor(bufB, C1, row0, kc * 32);
      v16h w = load_b_packed(W1, C2 / 16, kc, ct);
      acc = wmma_f16(a, w, acc);
    }
    int n = ct * 16 + nl;
    float bb = b1[n], sc = g1[n] * BN_SCALE, bt = be1[n];
#pragma unroll
    for (int r = 0; r < 8; ++r) {
      int mm = row0 + r + half * 8;
      float v = (acc[r] + bb) * sc + bt;
      bufA[mm * CA + n] = (_Float16)(v > 0.0f ? v : 0.0f);
    }
  }
  __syncthreads();

  // layer 2: bufA(K=C2) -> masked max over 64 neighbor rows (bias after max)
  unsigned long long mk = mask[bm];
  for (int ct = 0; ct < C3 / 16; ++ct) {
    v8f acc = {};
    for (int kc = 0; kc < C2 / 32; ++kc) {
      v16h a = load_a_rowmajor(bufA, CA, row0, kc * 32);
      v16h w = load_b_packed(W2, C3 / 16, kc, ct);
      acc = wmma_f16(a, w, acc);
    }
    float vm = -3.0e38f;
#pragma unroll
    for (int r = 0; r < 8; ++r) {
      int mm = row0 + r + half * 8;
      if ((mk >> mm) & 1ull) vm = fmaxf(vm, acc[r]);
    }
    float o = __shfl_xor(vm, 16, 32);   // fold the two half-wave row groups
    vm = fmaxf(vm, o);
    if (half == 0) pool[wave * C3 + ct * 16 + nl] = vm;
  }
  __syncthreads();
  for (int c = tid; c < C3; c += 128) {
    float v = fmaxf(fmaxf(pool[c], pool[C3 + c]),
                    fmaxf(pool[2 * C3 + c], pool[3 * C3 + c])) + b2[c];
    xout[bm * C3 + c] = v;
  }
}

// ---------------------------------------------------------------------------
// Generic WMMA GEMM for the SA3 chain (A f16 row-major, W packed; L2-resident)
// block = 4 waves; each wave: 16 rows x 64 cols, K-outer with 4 accumulators
// so each A fragment feeds 4 WMMAs.
// ---------------------------------------------------------------------------
template <int KPAD, int N, int EPI>   // EPI 0: bias+BN+ReLU -> f16, 1: bias -> f32
__global__ void __launch_bounds__(128)
gemm_kernel(const _Float16* __restrict__ A, const _Float16* __restrict__ Wp,
            const float* __restrict__ bias, const float* __restrict__ gamma,
            const float* __restrict__ beta,
            _Float16* __restrict__ outH, float* __restrict__ outF) {
  int wave = threadIdx.x >> 5, lane = threadIdx.x & 31;
  int half = lane >> 4, nl = lane & 15;
  int m0 = blockIdx.x * 64 + wave * 16;
  int nt0 = blockIdx.y * 4;

  v8f acc[4] = {{}, {}, {}, {}};
  for (int kc = 0; kc < KPAD / 32; ++kc) {
    v16h a = load_a_rowmajor(A, KPAD, m0, kc * 32);
#pragma unroll
    for (int ct = 0; ct < 4; ++ct) {
      v16h w = load_b_packed(Wp, N / 16, kc, nt0 + ct);
      acc[ct] = wmma_f16(a, w, acc[ct]);
    }
  }
#pragma unroll
  for (int ct = 0; ct < 4; ++ct) {
    int n = (nt0 + ct) * 16 + nl;
    if (EPI == 0) {
      float bb = bias[n], sc = gamma[n] * BN_SCALE, bt = beta[n];
#pragma unroll
      for (int r = 0; r < 8; ++r) {
        int m = m0 + r + half * 8;
        float v = (acc[ct][r] + bb) * sc + bt;
        outH[m * N + n] = (_Float16)(v > 0.0f ? v : 0.0f);
      }
    } else {
      float bb = bias[n];
#pragma unroll
      for (int r = 0; r < 8; ++r) {
        int m = m0 + r + half * 8;
        outF[m * N + n] = acc[ct][r] + bb;
      }
    }
  }
}

// ---------------------------------------------------------------------------
// Build SA3 input A0 [2048][288] = [x2 (256) | p2 (3) | pad]
// ---------------------------------------------------------------------------
__global__ void stage_sa3_kernel(const float* __restrict__ x2,
                                 const float* __restrict__ p2,
                                 _Float16* __restrict__ A0) {
  int idx = blockIdx.x * blockDim.x + threadIdx.x;
  if (idx >= 2048 * 288) return;
  int row = idx / 288, c = idx % 288;
  float v;
  if (c < 256)      v = x2[row * 256 + c];
  else if (c < 259) v = p2[row * 3 + (c - 256)];
  else              v = 0.0f;
  A0[idx] = (_Float16)v;
}

// global max pool over 128 points per cloud (coalesced over columns)
__global__ void pool3_kernel(const float* __restrict__ G3, float* __restrict__ g) {
  int b = blockIdx.x, c = threadIdx.x;  // blockDim = 1024
  float v = -3.4e38f;
  for (int m = 0; m < 128; ++m) v = fmaxf(v, G3[(b * 128 + m) * 1024 + c]);
  g[b * 1024 + c] = v;
}

// ---------------------------------------------------------------------------
// Head: 1024->512->256->10 + log_softmax, per cloud (tiny, plain f32)
// ---------------------------------------------------------------------------
__global__ void head_kernel(const float* __restrict__ g,
                            const float* __restrict__ W0, const float* __restrict__ b0,
                            const float* __restrict__ W1, const float* __restrict__ b1,
                            const float* __restrict__ W2, const float* __restrict__ b2,
                            float* __restrict__ out) {
  __shared__ float sg[1024], h1[512], h2[256], z[10];
  int b = blockIdx.x, tid = threadIdx.x;  // blockDim = 256
  for (int i = tid; i < 1024; i += 256) sg[i] = g[b * 1024 + i];
  __syncthreads();
  for (int c = tid; c < 512; c += 256) {
    float a = b0[c];
    for (int k = 0; k < 1024; ++k) a += sg[k] * W0[k * 512 + c];
    h1[c] = fmaxf(a, 0.0f);
  }
  __syncthreads();
  for (int c = tid; c < 256; c += 256) {
    float a = b1[c];
    for (int k = 0; k < 512; ++k) a += h1[k] * W1[k * 256 + c];
    h2[c] = fmaxf(a, 0.0f);
  }
  __syncthreads();
  if (tid < 10) {
    float a = b2[tid];
    for (int k = 0; k < 256; ++k) a += h2[k] * W2[k * 10 + tid];
    z[tid] = a;
  }
  __syncthreads();
  if (tid == 0) {
    float mx = z[0];
    for (int c = 1; c < 10; ++c) mx = fmaxf(mx, z[c]);
    float s = 0.0f;
    for (int c = 0; c < 10; ++c) s += expf(z[c] - mx);
    float l = logf(s);
    for (int c = 0; c < 10; ++c) out[b * 10 + c] = z[c] - mx - l;
  }
}

// ---------------------------------------------------------------------------
extern "C" void kernel_launch(void* const* d_in, const int* in_sizes, int n_in,
                              void* d_out, int out_size, void* d_ws, size_t ws_size,
                              hipStream_t stream) {
  (void)in_sizes; (void)n_in; (void)out_size; (void)ws_size;
  const float* pos = (const float*)d_in[0];
  // sa1 layers
  const float *W10f = (const float*)d_in[2],  *b10 = (const float*)d_in[3];
  const float *g10  = (const float*)d_in[4],  *be10 = (const float*)d_in[5];
  const float *W11f = (const float*)d_in[6],  *b11 = (const float*)d_in[7];
  const float *g11  = (const float*)d_in[8],  *be11 = (const float*)d_in[9];
  const float *W12f = (const float*)d_in[10], *b12 = (const float*)d_in[11];
  // sa2
  const float *W20f = (const float*)d_in[12], *b20 = (const float*)d_in[13];
  const float *g20  = (const float*)d_in[14], *be20 = (const float*)d_in[15];
  const float *W21f = (const float*)d_in[16], *b21 = (const float*)d_in[17];
  const float *g21  = (const float*)d_in[18], *be21 = (const float*)d_in[19];
  const float *W22f = (const float*)d_in[20], *b22 = (const float*)d_in[21];
  // sa3
  const float *W30f = (const float*)d_in[22], *b30 = (const float*)d_in[23];
  const float *g30  = (const float*)d_in[24], *be30 = (const float*)d_in[25];
  const float *W31f = (const float*)d_in[26], *b31 = (const float*)d_in[27];
  const float *g31  = (const float*)d_in[28], *be31 = (const float*)d_in[29];
  const float *W32f = (const float*)d_in[30], *b32 = (const float*)d_in[31];
  // head
  const float *hW0 = (const float*)d_in[32], *hb0 = (const float*)d_in[33];
  const float *hW1 = (const float*)d_in[34], *hb1 = (const float*)d_in[35];
  const float *hW2 = (const float*)d_in[36], *hb2 = (const float*)d_in[37];

  char* ws = (char*)d_ws;
  size_t off = 0;
  auto alloc = [&](size_t bytes) -> char* {
    char* p = ws + off;
    off += (bytes + 255) & ~(size_t)255;
    return p;
  };

  _Float16* wf = (_Float16*)alloc((size_t)813056 * 2);
  _Float16 *W10 = wf + 0,      *W11 = wf + 2048,   *W12 = wf + 6144;
  _Float16 *W20 = wf + 14336,  *W21 = wf + 34816,  *W22 = wf + 51200;
  _Float16 *W30 = wf + 83968,  *W31 = wf + 157696, *W32 = wf + 288768;

  int*   idx1  = (int*)  alloc(16 * 512 * 4);
  float* p1    = (float*)alloc(16 * 512 * 3 * 4);
  int*   nbr1  = (int*)  alloc((size_t)16 * 512 * 64 * 4);
  unsigned long long* mask1 = (unsigned long long*)alloc(16 * 512 * 8);
  float* x1    = (float*)alloc((size_t)16 * 512 * 128 * 4);
  int*   idx2  = (int*)  alloc(16 * 128 * 4);
  float* p2    = (float*)alloc(16 * 128 * 3 * 4);
  int*   nbr2  = (int*)  alloc((size_t)16 * 128 * 64 * 4);
  unsigned long long* mask2 = (unsigned long long*)alloc(16 * 128 * 8);
  float* x2    = (float*)alloc((size_t)16 * 128 * 256 * 4);
  _Float16* A0 = (_Float16*)alloc((size_t)2048 * 288 * 2);
  _Float16* A1 = (_Float16*)alloc((size_t)2048 * 256 * 2);
  _Float16* A2 = (_Float16*)alloc((size_t)2048 * 512 * 2);
  float* G3    = (float*)alloc((size_t)2048 * 1024 * 4);
  float* gbuf  = (float*)alloc(16 * 1024 * 4);

  auto pack = [&](const float* s, _Float16* d, int K, int N, int KPAD) {
    int tot = KPAD * N;
    pack_b_kernel<<<(tot + 255) / 256, 256, 0, stream>>>(s, d, K, N, KPAD);
  };
  pack(W10f, W10, 3, 64, 32);
  pack(W11f, W11, 64, 64, 64);
  pack(W12f, W12, 64, 128, 64);
  pack(W20f, W20, 131, 128, 160);
  pack(W21f, W21, 128, 128, 128);
  pack(W22f, W22, 128, 256, 128);
  pack(W30f, W30, 259, 256, 288);
  pack(W31f, W31, 256, 512, 256);
  pack(W32f, W32, 512, 1024, 512);

  // ---- SA1: FPS 1024->512, KNN r=0.2, MLP [3,64,64,128] + max-pool
  fps_kernel<<<16, 1024, 0, stream>>>(pos, 1024, 512, idx1, p1);
  knn_kernel<<<16 * 512, 128, 0, stream>>>(pos, 1024, p1, 512, 0.04f, nbr1, mask1);
  sa_kernel<0, 32, 64, 64, 128><<<16 * 512, 128, 0, stream>>>(
      pos, (const float*)nullptr, p1, nbr1, mask1, 1024, 512,
      W10, W11, W12, b10, g10, be10, b11, g11, be11, b12, x1);

  // ---- SA2: FPS 512->128, KNN r=0.4, MLP [131,128,128,256] + max-pool
  fps_kernel<<<16, 512, 0, stream>>>(p1, 512, 128, idx2, p2);
  knn_kernel<<<16 * 128, 128, 0, stream>>>(p1, 512, p2, 128, 0.16f, nbr2, mask2);
  sa_kernel<128, 160, 128, 128, 256><<<16 * 128, 128, 0, stream>>>(
      p1, x1, p2, nbr2, mask2, 512, 128,
      W20, W21, W22, b20, g20, be20, b21, g21, be21, b22, x2);

  // ---- SA3 (global): [259,256,512,1024] GEMM chain + global max-pool
  stage_sa3_kernel<<<(2048 * 288 + 255) / 256, 256, 0, stream>>>(x2, p2, A0);
  gemm_kernel<288, 256, 0><<<dim3(32, 4), 128, 0, stream>>>(
      A0, W30, b30, g30, be30, A1, (float*)nullptr);
  gemm_kernel<256, 512, 0><<<dim3(32, 8), 128, 0, stream>>>(
      A1, W31, b31, g31, be31, A2, (float*)nullptr);
  gemm_kernel<512, 1024, 1><<<dim3(32, 16), 128, 0, stream>>>(
      A2, W32, b32, (const float*)nullptr, (const float*)nullptr,
      (_Float16*)nullptr, G3);
  pool3_kernel<<<16, 1024, 0, stream>>>(G3, gbuf);

  // ---- head + log_softmax
  head_kernel<<<16, 256, 0, stream>>>(gbuf, hW0, hb0, hW1, hb1, hW2, hb2,
                                      (float*)d_out);
}
